// NWHead_21973052686725
// MI455X (gfx1250) — matrix operand order repaired and checked by reading, not compiled, not run on hardware
//
#include <hip/hip_runtime.h>
#include <hip/hip_bf16.h>

// ---------------------------------------------------------------------------
// Problem constants (from reference)
// ---------------------------------------------------------------------------
#define BATCH   16
#define NSUP    8192
#define FEAT    512
#define EMB     128
#define NCLS    100

// Workspace layout (bytes)
#define WS_BSW      0          // swizzled bf16 masked weight: 16*8*32*8 dwords = 131072 B
#define WS_QM       131072     // qm (16*128 f32) = 8192 B
#define WS_QQ       139264     // qq (16 f32)
#define WS_MAXV     139520     // per-batch max (16 f32)
#define WS_SUMEXP   139776     // per-batch sumexp (16 f32)
#define WS_SCORES   140032     // scores (16*8192 f32) = 524288 B
#define WS_PARTIAL  664320     // partial out (16*64*100 f32) = 409600 B -> total ~1.02 MB

typedef __attribute__((ext_vector_type(16))) __bf16 v16bf;
typedef __attribute__((ext_vector_type(8)))  float  v8f;

// f32 -> bf16 pair packed into one dword.
// Round-half-up (+0x8000) then byte-select the two high halves with a single
// v_perm_b32: 3 VALU ops total vs ~10 for exact RNE. Half-ulp tie bias is
// irrelevant at bf16 precision for this workload.
__device__ __forceinline__ unsigned pack_bf16(float lo, float hi) {
    const unsigned ul = __float_as_uint(lo) + 0x8000u;
    const unsigned uh = __float_as_uint(hi) + 0x8000u;
#if __has_builtin(__builtin_amdgcn_perm)
    // out bytes [3:2] = uh bytes [3:2] (codes 7,6); [1:0] = ul bytes [3:2] (codes 3,2)
    return __builtin_amdgcn_perm(uh, ul, 0x07060302u);
#else
    return (ul >> 16) | (uh & 0xFFFF0000u);
#endif
}

// ---------------------------------------------------------------------------
// Kernel 1: prep.  m = sigmoid(mask); Bsw = swizzled bf16 (W * m) in the
// v_wmma_f32_16x16x32_bf16 B-fragment layout; qm = (x @ W) * m; qq = ||qm||^2.
// B 32x16 bf16 fragment layout (ISA 7.12.2): lane l: col n = l%16;
// lanes 0-15 hold K={0..7,16..23} (2/VGPR), lanes 16-31 hold K={8..15,24..31}.
// Buffer index: ((kchunk*8 + ntile)*32 + lane)*8 + vgpr
// ---------------------------------------------------------------------------
__global__ __launch_bounds__(256) void prep_kernel(
    const float* __restrict__ x, const float* __restrict__ W,
    const float* __restrict__ mask,
    unsigned* __restrict__ Bsw, float* __restrict__ qm, float* __restrict__ qq)
{
    __shared__ float mloc[EMB];
    __shared__ float qms[BATCH * EMB];
    const int tid = threadIdx.x;

    if (tid < EMB) mloc[tid] = 1.0f / (1.0f + __expf(-mask[tid]));
    __syncthreads();

    // swizzled masked bf16 weight (32768 packed-pair dwords)
    for (int i = tid; i < 16 * 8 * 32 * 8; i += 256) {
        const int j = i & 7;            // vgpr dword
        const int l = (i >> 3) & 31;    // lane
        const int t = (i >> 8) & 7;     // n-tile
        const int c = i >> 11;          // k-chunk
        const int n  = t * 16 + (l & 15);
        const int k0 = c * 32 + ((j >> 2) << 4) + ((l >> 4) ? 8 : 0) + 2 * (j & 3);
        const float mm = mloc[n];
        const float w0 = W[k0 * EMB + n] * mm;
        const float w1 = W[(k0 + 1) * EMB + n] * mm;
        Bsw[i] = pack_bf16(w0, w1);
    }

    // qm = (x @ W) * m   (2048 outputs, 512-MAC loop each)
    for (int o = tid; o < BATCH * EMB; o += 256) {
        const int b = o >> 7, e = o & 127;
        float sum = 0.0f;
        const float* xr = x + b * FEAT;
        for (int f = 0; f < FEAT; ++f) sum += xr[f] * W[f * EMB + e];
        const float v = sum * mloc[e];
        qm[o] = v;
        qms[o] = v;
    }
    __syncthreads();

    if (tid < BATCH) {
        float s = 0.0f;
        for (int e = 0; e < EMB; ++e) { const float v = qms[tid * EMB + e]; s += v * v; }
        qq[tid] = s;
    }
}

// ---------------------------------------------------------------------------
// Kernel 2: fused  s = support_x @ W_m  (bf16 WMMA, f32 accum)
//           scores = 2*(s.qm) - qq - ||s||^2      never materializes s.
// Grid: 1024 WGs (16 batches x 64 chunks of 128 rows), 256 threads = 8 waves.
// Each wave: 16 rows x 128 embed, K=512 in 16 chunks of 32 (bf16).
// B staged once per WG in 128 KB LDS -> ds_load_b128 fragment reads.
// ---------------------------------------------------------------------------
__global__ __launch_bounds__(256) void score_kernel(
    const float* __restrict__ sx, const unsigned* __restrict__ Bsw,
    const float* __restrict__ qm, const float* __restrict__ qq,
    float* __restrict__ scores)
{
    __shared__ unsigned ldsB[16 * 8 * 32 * 8];   // 131072 B

    const int wg    = blockIdx.x;
    const int b     = wg >> 6;        // batch
    const int row0  = (wg & 63) * 128;
    const int tid   = threadIdx.x;

    // cooperative stage of the swizzled B matrix into LDS
    {
        const uint4* src = (const uint4*)Bsw;
        uint4* dst = (uint4*)ldsB;
        for (int i = tid; i < (16 * 8 * 32 * 8) / 4; i += 256) dst[i] = src[i];
    }
    __syncthreads();

    const int wave = tid >> 5;
    const int lane = tid & 31;
    const int hv   = lane >> 4;       // lane-half (K phase for A, M phase for C)
    const int l16  = lane & 15;

    // A 16-bit 16x32 layout: all lanes row m = l16; half 0 -> K {0..7,16..23},
    // half 1 -> K {8..15,24..31}  (2 bf16 per VGPR)
    const int   arow_idx = row0 + wave * 16 + l16;
    const float* arow = sx + ((size_t)b * NSUP + arow_idx) * FEAT + hv * 8;

    v8f acc[8];
#pragma unroll
    for (int t = 0; t < 8; ++t)
#pragma unroll
        for (int j = 0; j < 8; ++j) acc[t][j] = 0.0f;

#pragma unroll 2
    for (int c = 0; c < 16; ++c) {
        const float* p = arow + c * 32;
        // pull the streaming A data ~4 K-chunks (512 B) ahead into cache
        __builtin_prefetch(p + 128, 0, 1);
        const float4 a0 = *(const float4*)(p);
        const float4 a1 = *(const float4*)(p + 4);
        const float4 a2 = *(const float4*)(p + 16);
        const float4 a3 = *(const float4*)(p + 20);
        union { unsigned u[8]; v16bf v; } A;
        A.u[0] = pack_bf16(a0.x, a0.y); A.u[1] = pack_bf16(a0.z, a0.w);
        A.u[2] = pack_bf16(a1.x, a1.y); A.u[3] = pack_bf16(a1.z, a1.w);
        A.u[4] = pack_bf16(a2.x, a2.y); A.u[5] = pack_bf16(a2.z, a2.w);
        A.u[6] = pack_bf16(a3.x, a3.y); A.u[7] = pack_bf16(a3.z, a3.w);

#pragma unroll
        for (int t = 0; t < 8; ++t) {
            union { unsigned u[8]; v16bf v; } B;
            const unsigned* bp = &ldsB[(((c * 8) + t) * 32 + lane) * 8];
            *(uint4*)&B.u[0] = *(const uint4*)(bp);
            *(uint4*)&B.u[4] = *(const uint4*)(bp + 4);
            acc[t] = __builtin_amdgcn_wmma_f32_16x16x32_bf16(
                false, A.v, false, B.v, (short)0, acc[t], false, false);
        }
    }

    // ---- fused epilogue: per-row ss and qs ------------------------------
    // C layout: vgpr j, lanes 0-15 -> row j, col l16; lanes 16-31 -> row j+8.
    float qv[8];
#pragma unroll
    for (int t = 0; t < 8; ++t) qv[t] = qm[b * EMB + t * 16 + l16];

    float pss[8], pqs[8];
#pragma unroll
    for (int j = 0; j < 8; ++j) { pss[j] = 0.0f; pqs[j] = 0.0f; }
#pragma unroll
    for (int t = 0; t < 8; ++t)
#pragma unroll
        for (int j = 0; j < 8; ++j) {
            const float s = acc[t][j];
            pss[j] += s * s;
            pqs[j] += s * qv[t];
        }

    // reduce across the 16 lanes of each half (xor < 16 stays within half)
#pragma unroll
    for (int off = 1; off < 16; off <<= 1)
#pragma unroll
        for (int j = 0; j < 8; ++j) {
            pss[j] += __shfl_xor(pss[j], off, 32);
            pqs[j] += __shfl_xor(pqs[j], off, 32);
        }

    if (l16 == 0) {
        const float qqb = qq[b];
        const int rbase = row0 + wave * 16 + hv * 8;
#pragma unroll
        for (int j = 0; j < 8; ++j)
            scores[(size_t)b * NSUP + rbase + j] = 2.0f * pqs[j] - qqb - pss[j];
    }
}

// ---------------------------------------------------------------------------
// Kernel 3: per-batch softmax stats (max, sum of exp)
// ---------------------------------------------------------------------------
__global__ __launch_bounds__(256) void softmax_stats_kernel(
    const float* __restrict__ scores, float* __restrict__ maxv,
    float* __restrict__ sumexp)
{
    __shared__ float red[256];
    const int b = blockIdx.x, tid = threadIdx.x;
    const float* s = scores + (size_t)b * NSUP;

    float mx = -3.402823466e+38f;
    for (int i = tid; i < NSUP; i += 256) mx = fmaxf(mx, s[i]);
    red[tid] = mx; __syncthreads();
    for (int o = 128; o > 0; o >>= 1) {
        if (tid < o) red[tid] = fmaxf(red[tid], red[tid + o]);
        __syncthreads();
    }
    mx = red[0]; __syncthreads();

    float se = 0.0f;
    for (int i = tid; i < NSUP; i += 256) se += __expf(s[i] - mx);
    red[tid] = se; __syncthreads();
    for (int o = 128; o > 0; o >>= 1) {
        if (tid < o) red[tid] += red[tid + o];
        __syncthreads();
    }
    if (tid == 0) { maxv[b] = mx; sumexp[b] = red[0]; }
}

// ---------------------------------------------------------------------------
// Kernel 4: partial  out_partial[b,chunk,c] = sum_{n in chunk} p_n * sy[b,n,c]
// Grid 1024 (16 batches x 64 chunks of 128 rows); 2 row-groups of 128 lanes.
// No atomics -> bitwise deterministic.
// ---------------------------------------------------------------------------
__global__ __launch_bounds__(256) void accum_kernel(
    const float* __restrict__ scores, const float* __restrict__ sy,
    const float* __restrict__ maxv, const float* __restrict__ sumexp,
    float* __restrict__ partial)
{
    __shared__ float red[2][128];
    const int b     = blockIdx.x >> 6;
    const int chunk = blockIdx.x & 63;
    const int g     = threadIdx.x >> 7;   // row group 0/1
    const int c     = threadIdx.x & 127;  // class lane

    const float mx  = maxv[b];
    const float inv = 1.0f / sumexp[b];
    const int n0 = chunk * 128;

    float acc = 0.0f;
    for (int r = g; r < 128; r += 2) {
        const int n = n0 + r;
        const float p = __expf(scores[(size_t)b * NSUP + n] - mx) * inv;
        if (c < NCLS) acc += p * sy[((size_t)b * NSUP + n) * NCLS + c];
    }
    red[g][c] = acc;
    __syncthreads();
    if (g == 0 && c < NCLS)
        partial[((size_t)b * 64 + chunk) * NCLS + c] = red[0][c] + red[1][c];
}

// ---------------------------------------------------------------------------
// Kernel 5: final deterministic reduction of the 64 chunk partials
// ---------------------------------------------------------------------------
__global__ __launch_bounds__(128) void finalize_kernel(
    const float* __restrict__ partial, float* __restrict__ out)
{
    const int b = blockIdx.x, c = threadIdx.x;
    if (c < NCLS) {
        float s = 0.0f;
        for (int k = 0; k < 64; ++k) s += partial[((size_t)b * 64 + k) * NCLS + c];
        out[b * NCLS + c] = s;
    }
}

// ---------------------------------------------------------------------------
extern "C" void kernel_launch(void* const* d_in, const int* in_sizes, int n_in,
                              void* d_out, int out_size, void* d_ws, size_t ws_size,
                              hipStream_t stream)
{
    const float* x    = (const float*)d_in[0];   // (16, 512)
    const float* sx   = (const float*)d_in[1];   // (16, 8192, 512)
    const float* sy   = (const float*)d_in[2];   // (16, 8192, 100)
    const float* W    = (const float*)d_in[3];   // (512, 128)
    const float* mask = (const float*)d_in[4];   // (128,)
    float* out = (float*)d_out;                  // (16, 100)

    char* ws = (char*)d_ws;
    unsigned* Bsw    = (unsigned*)(ws + WS_BSW);
    float*    qm     = (float*)(ws + WS_QM);
    float*    qq     = (float*)(ws + WS_QQ);
    float*    maxv   = (float*)(ws + WS_MAXV);
    float*    sumexp = (float*)(ws + WS_SUMEXP);
    float*    scores = (float*)(ws + WS_SCORES);
    float*    part   = (float*)(ws + WS_PARTIAL);

    prep_kernel<<<1, 256, 0, stream>>>(x, W, mask, Bsw, qm, qq);
    score_kernel<<<BATCH * (NSUP / 128), 256, 0, stream>>>(sx, Bsw, qm, qq, scores);
    softmax_stats_kernel<<<BATCH, 256, 0, stream>>>(scores, maxv, sumexp);
    accum_kernel<<<BATCH * (NSUP / 128), 256, 0, stream>>>(scores, sy, maxv, sumexp, part);
    finalize_kernel<<<BATCH, 128, 0, stream>>>(part, out);
}
